// subDDG_Attention_14233521619105
// MI455X (gfx1250) — matrix operand order repaired and checked by reading, not compiled, not run on hardware
//
#include <hip/hip_runtime.h>

// ---------------------------------------------------------------------------
// CDNA5 (gfx1250) fused windowed-attention block:
//   y = proj2( attn( proj1(x) ) )   with bf16 WMMA + async-to-LDS staging.
// ---------------------------------------------------------------------------

typedef __attribute__((ext_vector_type(16))) __bf16 v16bf;
typedef __attribute__((ext_vector_type(8)))  float  v8f;

union Frag {                 // one 16x32 bf16 WMMA operand per lane
    uint4 q[2];              // 32 bytes
    v16bf v;
};

__device__ __forceinline__ unsigned short f2bf(float f) {
    unsigned u = __builtin_bit_cast(unsigned, f);
    u += 0x7fffu + ((u >> 16) & 1u);       // round-to-nearest-even
    return (unsigned short)(u >> 16);
}

__device__ __forceinline__ v8f wmma_bf16(v16bf a, v16bf b, v8f c) {
    // D = A(16x32 bf16) * B(32x16 bf16) + C(16x16 f32)
    return __builtin_amdgcn_wmma_f32_16x16x32_bf16(
        /*neg_a=*/false, a, /*neg_b=*/false, b,
        /*c_mod=*/(short)0, c, /*reuse_a=*/false, /*reuse_b=*/false);
}

// Async copy of 16 bytes global -> LDS (GV mode, tracked by ASYNCcnt).
__device__ __forceinline__ void async_g2l_b128(const void* lds, const void* gptr) {
    unsigned loff = (unsigned)(unsigned long long)lds;      // low 32b = LDS offset
    unsigned long long ga = (unsigned long long)gptr;
    asm volatile("global_load_async_to_lds_b128 %0, %1, off"
                 :: "v"(loff), "v"(ga) : "memory");
}

// ---------------------------------------------------------------------------
// Prep: f32 [R][C] weight -> bf16 [C][R] (transposed, so GEMM B tiles stage
// exactly like A tiles: contiguous along the reduction dim).
// ---------------------------------------------------------------------------
__global__ void convT_f32_to_bf16(const float* __restrict__ W,
                                  unsigned short* __restrict__ WT,
                                  int R, int C) {
    int idx = blockIdx.x * blockDim.x + threadIdx.x;
    if (idx >= R * C) return;
    int r = idx / C, c = idx - r * C;
    WT[(long)c * R + r] = f2bf(W[idx]);
}

// Prep: relative-position bias -> padded [8][64][64] f32 with -1e30 pads.
__global__ void build_bias(const float* __restrict__ table,   // [169][8]
                           const int*   __restrict__ relidx,  // [49][49]
                           float* __restrict__ biasM) {       // [8][64][64]
    int idx = blockIdx.x * blockDim.x + threadIdx.x;
    if (idx >= 8 * 64 * 64) return;
    int h = idx >> 12, q = (idx >> 6) & 63, k = idx & 63;
    float v;
    if (q < 49 && k < 49) v = table[relidx[q * 49 + k] * 8 + h];
    else                  v = (k == 0) ? 0.0f : -1e30f;   // pads kill softmax
    biasM[idx] = v;
}

// ---------------------------------------------------------------------------
// Tiled WMMA GEMM:  Out[M][N] = A[M][K] * BT[N][K]^T + bias[N]
//   block = 256 thr (8 waves), block tile 128x256, k-step 32
//   wave grid 2(M)x4(N): each wave -> 64x64 = 4x4 WMMA accumulators
//   Double-buffered async-to-LDS staging for bf16 tiles (ASYNCcnt);
//   f32 A (proj1) converts through a register pipeline instead.
// ---------------------------------------------------------------------------
template <int N, int K, bool A_F32, bool OUT_F32>
__launch_bounds__(256)
__global__ void gemm_wmma(const void* __restrict__ Aptr,
                          const unsigned short* __restrict__ BT,
                          const float* __restrict__ bias,
                          void* __restrict__ Out) {
    constexpr int LDA = 40;                 // bf16 units; 80B rows, 16B aligned
    constexpr int GRP = A_F32 ? 4 : 6;      // async ops issued per thread/step
    __shared__ __align__(16) unsigned short As[2][128 * LDA];
    __shared__ __align__(16) unsigned short Bs[2][256 * LDA];

    const int tid  = threadIdx.x;
    const int wave = tid >> 5, lane = tid & 31;
    const int lm   = lane & 15, half = lane >> 4;
    const int wm   = wave >> 2;             // 0..1  (M)
    const int wn   = wave & 3;              // 0..3  (N)

    const long row0 = (long)blockIdx.x * 128;
    const int  col0 = blockIdx.y * 256;

    const int aRow = tid >> 1;              // 0..127
    const int aSeg = (tid & 1) * 16;        // 0 / 16 along K

    const float*          Af = (const float*)Aptr;
    const unsigned short* Ah = (const unsigned short*)Aptr;

    float4 arf[4];                          // A f32 register pipeline (proj1)

    auto loadAregs = [&](int k0) {
        const float4* s = (const float4*)(Af + (row0 + aRow) * K + k0 + aSeg);
        arf[0] = s[0]; arf[1] = s[1]; arf[2] = s[2]; arf[3] = s[3];
    };
    auto storeAregs = [&]() {               // convert + store into As[0]
        unsigned pk[8];
#pragma unroll
        for (int v = 0; v < 4; ++v) {
            float4 f = arf[v];
            pk[2 * v + 0] = (unsigned)f2bf(f.x) | ((unsigned)f2bf(f.y) << 16);
            pk[2 * v + 1] = (unsigned)f2bf(f.z) | ((unsigned)f2bf(f.w) << 16);
        }
        uint4* d = (uint4*)&As[0][aRow * LDA + aSeg];
        d[0] = make_uint4(pk[0], pk[1], pk[2], pk[3]);
        d[1] = make_uint4(pk[4], pk[5], pk[6], pk[7]);
    };
    auto issueAsync = [&](int buf, int k0) {
        const unsigned short* gb = BT + (long)(col0 + tid) * K + k0;
#pragma unroll
        for (int i = 0; i < 4; ++i)         // B row of 32 bf16 = 4 x b128
            async_g2l_b128(&Bs[buf][tid * LDA + 8 * i], gb + 8 * i);
        if constexpr (!A_F32) {
            const unsigned short* ga = Ah + (row0 + aRow) * K + k0 + aSeg;
#pragma unroll
            for (int i = 0; i < 2; ++i)     // A half-row of 16 bf16 = 2 x b128
                async_g2l_b128(&As[buf][aRow * LDA + aSeg + 8 * i], ga + 8 * i);
        }
    };

    v8f acc[4][4];
#pragma unroll
    for (int m = 0; m < 4; ++m)
#pragma unroll
        for (int n = 0; n < 4; ++n)
            acc[m][n] = (v8f){0.f, 0.f, 0.f, 0.f, 0.f, 0.f, 0.f, 0.f};

    if constexpr (A_F32) loadAregs(0);
    issueAsync(0, 0);

    constexpr int STEPS = K / 32;
    for (int s = 0; s < STEPS; ++s) {
        const int k0  = s * 32;
        const int cur = s & 1;
        const bool hasNext = (s + 1 < STEPS);

        if constexpr (A_F32) storeAregs();
        if (hasNext) {
            if constexpr (A_F32) loadAregs(k0 + 32);
            issueAsync(cur ^ 1, k0 + 32);   // other buffer, overlaps compute
        }
        // wait for step-s group (in-order); leave the just-issued group inflight
        if (hasNext) asm volatile("s_wait_asynccnt %0" :: "i"(GRP) : "memory");
        else         asm volatile("s_wait_asynccnt 0" ::: "memory");
        __syncthreads();

        const unsigned short* Ab = As[A_F32 ? 0 : cur];
        const unsigned short* Bb = Bs[cur];

        Frag af[4];
#pragma unroll
        for (int m = 0; m < 4; ++m) {       // A: interleaved K per half-wave
            int r = wm * 64 + m * 16 + lm;
            af[m].q[0] = *(const uint4*)&Ab[r * LDA + half * 8];
            af[m].q[1] = *(const uint4*)&Ab[r * LDA + 16 + half * 8];
        }
#pragma unroll
        for (int n = 0; n < 4; ++n) {       // B: interleaved K per half-wave
            Frag bf;
            int c = wn * 64 + n * 16 + lm;
            const unsigned short* sp = &Bb[c * LDA];
            bf.q[0] = *(const uint4*)(sp + half * 8);
            bf.q[1] = *(const uint4*)(sp + 16 + half * 8);
#pragma unroll
            for (int m = 0; m < 4; ++m)
                acc[m][n] = wmma_bf16(af[m].v, bf.v, acc[m][n]);
        }
        __syncthreads();                    // reads done before buffer reuse
    }

    // ---- epilogue: +bias, store (f32 or bf16) ----
#pragma unroll
    for (int n = 0; n < 4; ++n) {
        int gc = col0 + wn * 64 + n * 16 + lm;
        float bv = bias[gc];
#pragma unroll
        for (int m = 0; m < 4; ++m) {
            long gr = row0 + wm * 64 + m * 16 + half * 8;
#pragma unroll
            for (int j = 0; j < 8; ++j) {
                float v = acc[m][n][j] + bv;
                if constexpr (OUT_F32)
                    ((float*)Out)[(gr + j) * N + gc] = v;
                else
                    ((unsigned short*)Out)[(gr + j) * N + gc] = f2bf(v);
            }
        }
    }
}

// ---------------------------------------------------------------------------
// Fused attention: one block per window (B_=2048), 4 waves, 2 heads/wave.
// L=49 padded to 64. S = (Q*scale)K^T + bias, softmax in registers,
// P staged bf16 in LDS, O = P*V via WMMA.
// QKV layout: [B*49][768] bf16, cols = s*256 + h*32 + d.
// ---------------------------------------------------------------------------
__launch_bounds__(128)
__global__ void attn_fused(const unsigned short* __restrict__ QKV,
                           const float* __restrict__ biasM, // [8][64][64]
                           unsigned short* __restrict__ O)  // [B*49][256]
{
    constexpr int LDP = 72;                 // bf16 units; 144B rows (16B align)
    __shared__ __align__(16) unsigned short Psh[4][64 * LDP];

    const int tid  = threadIdx.x;
    const int wave = tid >> 5, lane = tid & 31;
    const int lm   = lane & 15, half = lane >> 4;
    const long rowbase = (long)blockIdx.x * 49;
    const float scale  = 0.17677669529663687f;   // 1/sqrt(32)

    for (int hi = 0; hi < 2; ++hi) {
        const int h = wave + hi * 4;

        // Prefetch ALL Q and K fragments, then run the 16 WMMAs in one burst.
        Frag kf[4], qf[4];
#pragma unroll
        for (int n = 0; n < 4; ++n) {       // K: B operand, lane col = token
            int t = n * 16 + lm;
            if (t < 49) {
                const unsigned short* s = QKV + (rowbase + t) * 768 +
                                                256 + h * 32;
                kf[n].q[0] = *(const uint4*)(s + half * 8);
                kf[n].q[1] = *(const uint4*)(s + 16 + half * 8);
            } else {
                kf[n].q[0] = make_uint4(0, 0, 0, 0);
                kf[n].q[1] = make_uint4(0, 0, 0, 0);
            }
        }
#pragma unroll
        for (int m = 0; m < 4; ++m) {       // Q: A operand, interleaved d
            int q = m * 16 + lm;
            if (q < 49) {
                const unsigned short* b = QKV + (rowbase + q) * 768 + h * 32;
                qf[m].q[0] = *(const uint4*)(b + half * 8);
                qf[m].q[1] = *(const uint4*)(b + 16 + half * 8);
            } else {
                qf[m].q[0] = make_uint4(0, 0, 0, 0);
                qf[m].q[1] = make_uint4(0, 0, 0, 0);
            }
        }

        v8f S[4][4];
        const v8f z = (v8f){0.f, 0.f, 0.f, 0.f, 0.f, 0.f, 0.f, 0.f};
#pragma unroll
        for (int m = 0; m < 4; ++m)
#pragma unroll
            for (int n = 0; n < 4; ++n)
                S[m][n] = wmma_bf16(qf[m].v, kf[n].v, z);

        // softmax per row (row lives in one 16-lane half of C layout)
#pragma unroll
        for (int m = 0; m < 4; ++m) {
#pragma unroll
            for (int j = 0; j < 8; ++j) {
                int q = m * 16 + j + half * 8;
                float x[4];
#pragma unroll
                for (int n = 0; n < 4; ++n)
                    x[n] = S[m][n][j] * scale +
                           biasM[h * 4096 + q * 64 + n * 16 + lm];
                float mx = fmaxf(fmaxf(x[0], x[1]), fmaxf(x[2], x[3]));
#pragma unroll
                for (int d = 1; d < 16; d <<= 1)
                    mx = fmaxf(mx, __shfl_xor(mx, d, 32));
                float e[4], sum = 0.f;
#pragma unroll
                for (int n = 0; n < 4; ++n) { e[n] = __expf(x[n] - mx); sum += e[n]; }
#pragma unroll
                for (int d = 1; d < 16; d <<= 1)
                    sum += __shfl_xor(sum, d, 32);
                float inv = 1.0f / sum;
#pragma unroll
                for (int n = 0; n < 4; ++n)
                    Psh[wave][q * LDP + n * 16 + lm] = f2bf(e[n] * inv);
            }
        }
        __syncthreads();

        // O = P(64x64) * V(64x32): 2 d-tiles, 4 m-tiles, 2 k-steps
#pragma unroll
        for (int n2 = 0; n2 < 2; ++n2) {
            Frag vf[2];
#pragma unroll
            for (int s = 0; s < 2; ++s) {   // B: lane col = d, elems = token
                unsigned pk[8];
#pragma unroll
                for (int i = 0; i < 8; ++i) {
                    int t0 = s * 32 + (i >> 2) * 16 + half * 8 + 2 * (i & 3);
                    unsigned short lo = (t0 < 49)
                        ? QKV[(rowbase + t0) * 768 + 512 + h * 32 + n2 * 16 + lm]
                        : (unsigned short)0;
                    unsigned short hi2 = (t0 + 1 < 49)
                        ? QKV[(rowbase + t0 + 1) * 768 + 512 + h * 32 + n2 * 16 + lm]
                        : (unsigned short)0;
                    pk[i] = (unsigned)lo | ((unsigned)hi2 << 16);
                }
                vf[s].q[0] = make_uint4(pk[0], pk[1], pk[2], pk[3]);
                vf[s].q[1] = make_uint4(pk[4], pk[5], pk[6], pk[7]);
            }
#pragma unroll
            for (int m = 0; m < 4; ++m) {
                v8f accO = (v8f){0.f, 0.f, 0.f, 0.f, 0.f, 0.f, 0.f, 0.f};
#pragma unroll
                for (int s = 0; s < 2; ++s) {
                    Frag pf;
                    const unsigned short* b =
                        &Psh[wave][(m * 16 + lm) * LDP + s * 32];
                    pf.q[0] = *(const uint4*)(b + half * 8);
                    pf.q[1] = *(const uint4*)(b + 16 + half * 8);
                    accO = wmma_bf16(pf.v, vf[s].v, accO);
                }
#pragma unroll
                for (int j = 0; j < 8; ++j) {
                    int q = m * 16 + j + half * 8;
                    if (q < 49)
                        O[(rowbase + q) * 256 + h * 32 + n2 * 16 + lm] =
                            f2bf(accO[j]);
                }
            }
        }
        __syncthreads();
    }
}

// ---------------------------------------------------------------------------
// Host launcher
// ---------------------------------------------------------------------------
extern "C" void kernel_launch(void* const* d_in, const int* in_sizes, int n_in,
                              void* d_out, int out_size, void* d_ws, size_t ws_size,
                              hipStream_t stream) {
    const float* x     = (const float*)d_in[0];   // [2048*49][1024]
    const float* w1    = (const float*)d_in[1];   // [1024][256]
    const float* b1    = (const float*)d_in[2];   // [256]
    const float* wqkv  = (const float*)d_in[3];   // [256][768]
    const float* bqkv  = (const float*)d_in[4];   // [768]
    const float* w2    = (const float*)d_in[5];   // [256][1024]
    const float* b2    = (const float*)d_in[6];   // [1024]
    const float* table = (const float*)d_in[7];   // [169][8]
    const int*   relix = (const int*)d_in[8];     // [49][49]

    constexpr int M = 2048 * 49;                  // 100352 = 784 * 128

    char* ws = (char*)d_ws;
    unsigned short* w1T   = (unsigned short*)(ws + 0);          // 512 KB
    unsigned short* wqkvT = (unsigned short*)(ws + 524288);     // 384 KB
    unsigned short* w2T   = (unsigned short*)(ws + 917504);     // 512 KB
    float*          biasM = (float*)(ws + 1441792);             // 128 KB
    unsigned short* Hbuf  = (unsigned short*)(ws + 1572864);    // 49.0 MB
    unsigned short* QKVb  = (unsigned short*)(ws + 52953088);   // 147.0 MB
    unsigned short* Obuf  = (unsigned short*)(ws + 207093760);  // 49.0 MB

    // --- prep: weight transpose+convert, bias matrix ---
    convT_f32_to_bf16<<<(1024 * 256 + 255) / 256, 256, 0, stream>>>(w1, w1T, 1024, 256);
    convT_f32_to_bf16<<<(256 * 768 + 255) / 256, 256, 0, stream>>>(wqkv, wqkvT, 256, 768);
    convT_f32_to_bf16<<<(256 * 1024 + 255) / 256, 256, 0, stream>>>(w2, w2T, 256, 1024);
    build_bias<<<(8 * 64 * 64 + 255) / 256, 256, 0, stream>>>(table, relix, biasM);

    // --- proj1: H = x @ w1 + b1  (f32 in, bf16 out); x read exactly once ---
    gemm_wmma<256, 1024, true, false>
        <<<dim3(M / 128, 256 / 256), 256, 0, stream>>>(x, w1T, b1, Hbuf);

    // --- qkv: QKV = H @ wqkv + bqkv (bf16 in/out) ---
    gemm_wmma<768, 256, false, false>
        <<<dim3(M / 128, 768 / 256), 256, 0, stream>>>(Hbuf, wqkvT, bqkv, QKVb);

    // --- fused windowed attention ---
    attn_fused<<<2048, 128, 0, stream>>>(QKVb, biasM, Obuf);

    // --- proj2: y = O @ w2 + b2 (bf16 in, f32 out) ---
    gemm_wmma<1024, 256, false, true>
        <<<dim3(M / 128, 1024 / 256), 256, 0, stream>>>(Obuf, w2T, b2, (float*)d_out);
}
